// Model_23991687316177
// MI455X (gfx1250) — matrix operand order repaired and checked by reading, not compiled, not run on hardware
//
#include <hip/hip_runtime.h>
#include <hip/hip_bf16.h>
#include <stdint.h>

// ---------------------------------------------------------------------------
// MeshGraphNets on MI455X (gfx1250), compile-only target.
//   - bf16 activations/weights, f32 accumulation via v_wmma_f32_16x16x32_bf16
//   - fused 3-layer MLP kernels (gather -> L0 -> L1 -> L2 -> LN -> residual ->
//     scatter) so intermediates stay in LDS; weights stream from L2 (bf16,
//     ~18MB total, resident in 192MB L2).
//   - N=50000, E=300000 are both multiples of 16 => no tail handling.
// Input pointer order assumption: recursive dict-insertion order of
// setup_inputs(): velocity, mesh_pos, node_type, senders, receivers,
// node_mean, node_std, edge_mean, edge_std,
// enc_node{W0,b0,W1,b1,W2,b2,ln_g,ln_b}, enc_edge{...},
// proc_edge{stacked x15: W0,b0,W1,b1,W2,b2,ln_g,ln_b}, proc_node{...},
// dec{W0,b0,W1,b1,W2,b2}  => 47 pointers.
// ---------------------------------------------------------------------------

typedef __attribute__((ext_vector_type(16))) __bf16 v16bf;
typedef __attribute__((ext_vector_type(8)))  float  v8f;

#define LN_EPS 1e-5f

__device__ __forceinline__ v8f wmma_bf16(v16bf a, v16bf b, v8f c) {
  // D = A(16x32 bf16) x B(32x16 bf16) + C(16x16 f32)
  return __builtin_amdgcn_wmma_f32_16x16x32_bf16(false, a, false, b,
                                                 (short)0, c, false, false);
}

// A-operand (16x32, bf16) from LDS, row-major with element stride `stride`.
// Per ISA 7.12.2: lane<16 -> M=lane, K={kb..kb+7, kb+16..kb+23};
//                 lane>=16 -> M=lane-16, K offset +8.
__device__ __forceinline__ v16bf load_a(const __bf16* Xs, int stride, int kbase) {
  int lane = threadIdx.x & 31;
  const __bf16* p = Xs + (lane & 15) * stride + kbase + ((lane >> 4) << 3);
  union { v16bf v; uint4 q[2]; } u;
  u.q[0] = *(const uint4*)(p);       // K = kb .. kb+7
  u.q[1] = *(const uint4*)(p + 16);  // K = kb+16 .. kb+23
  return u.v;
}

// B-operand (32x16, bf16) from global. Weights pre-transposed: WT[n][k],
// row length K. lane<16 -> N=lane, K={kb..kb+7, kb+16..kb+23}; lane>=16 -> K offset +8.
__device__ __forceinline__ v16bf load_b(const __bf16* __restrict__ WT, int K,
                                        int ncolbase, int kbase) {
  int lane = threadIdx.x & 31;
  const __bf16* p = WT + (size_t)(ncolbase + (lane & 15)) * K + kbase + ((lane >> 4) << 3);
  union { v16bf v; uint4 q[2]; } u;
  u.q[0] = *(const uint4*)(p);
  u.q[1] = *(const uint4*)(p + 16);
  return u.v;
}

// One 16xK * Kx256 layer; each of the 8 waves owns 32 output columns.
__device__ __forceinline__ void gemm16(const __bf16* Xs, int xstride, int K,
                                       const __bf16* __restrict__ WT,
                                       v8f& acc0, v8f& acc1) {
  int wave = threadIdx.x >> 5;
  int c0 = wave * 32;
  v8f z = {0.f, 0.f, 0.f, 0.f, 0.f, 0.f, 0.f, 0.f};
  acc0 = z; acc1 = z;
  for (int kb = 0; kb < K; kb += 32) {
    v16bf a  = load_a(Xs, xstride, kb);
    v16bf b0 = load_b(WT, K, c0, kb);
    v16bf b1 = load_b(WT, K, c0 + 16, kb);
    acc0 = wmma_bf16(a, b0, acc0);
    acc1 = wmma_bf16(a, b1, acc1);
  }
}

// C/D layout (ISA 7.12.2): element j -> row = j + 8*(lane>=16), col = lane&15.
__device__ __forceinline__ void store_relu_bf16(v8f acc0, v8f acc1,
                                                const float* __restrict__ bias,
                                                __bf16* Hs, int hstride) {
  int wave = threadIdx.x >> 5, lane = threadIdx.x & 31;
  int rbase = (lane >> 4) << 3, n = lane & 15, c0 = wave * 32;
  float bv0 = bias[c0 + n], bv1 = bias[c0 + 16 + n];
#pragma unroll
  for (int j = 0; j < 8; ++j) {
    float x0 = fmaxf(acc0[j] + bv0, 0.f);
    float x1 = fmaxf(acc1[j] + bv1, 0.f);
    Hs[(rbase + j) * hstride + c0 + n]      = (__bf16)x0;
    Hs[(rbase + j) * hstride + c0 + 16 + n] = (__bf16)x1;
  }
}

__device__ __forceinline__ void store_f32(v8f acc0, v8f acc1,
                                          const float* __restrict__ bias,
                                          float* Fs, int fstride) {
  int wave = threadIdx.x >> 5, lane = threadIdx.x & 31;
  int rbase = (lane >> 4) << 3, n = lane & 15, c0 = wave * 32;
  float bv0 = bias[c0 + n], bv1 = bias[c0 + 16 + n];
#pragma unroll
  for (int j = 0; j < 8; ++j) {
    Fs[(rbase + j) * fstride + c0 + n]      = acc0[j] + bv0;
    Fs[(rbase + j) * fstride + c0 + 16 + n] = acc1[j] + bv1;
  }
}

// Row LayerNorm stats over 256 cols: 16 threads/row, 16 cols each,
// reduced with 16-lane shuffles (rows occupy contiguous 16-lane groups).
__device__ __forceinline__ void ln_stats(const float* Frow, int q,
                                         float& mu, float& rstd) {
  float s1 = 0.f, s2 = 0.f;
#pragma unroll
  for (int j = 0; j < 16; ++j) {
    float x = Frow[q * 16 + j];
    s1 += x; s2 += x * x;
  }
#pragma unroll
  for (int m = 1; m < 16; m <<= 1) {
    s1 += __shfl_xor(s1, m, 32);
    s2 += __shfl_xor(s2, m, 32);
  }
  mu = s1 * (1.f / 256.f);
  float var = s2 * (1.f / 256.f) - mu * mu;
  rstd = rsqrtf(var + LN_EPS);
}

// ------------------------------ prep kernels -------------------------------

// f32 W[k][n] (n=256) -> bf16 WT[n][k], K zero-padded to Kpad. batch=gridDim.y
__global__ void prep_wt(const float* __restrict__ src, __bf16* __restrict__ dst,
                        int K, int Kpad) {
  size_t s = blockIdx.y;
  src += s * (size_t)K * 256;
  dst += s * (size_t)256 * Kpad;
  int total = 256 * Kpad;
  for (int idx = blockIdx.x * blockDim.x + threadIdx.x; idx < total;
       idx += gridDim.x * blockDim.x) {
    int n = idx / Kpad, k = idx - n * Kpad;
    dst[idx] = (__bf16)((k < K) ? src[(size_t)k * 256 + n] : 0.f);
  }
}

__global__ void zero_f32(float* __restrict__ p, int n4) {
  int i = blockIdx.x * blockDim.x + threadIdx.x;
  if (i < n4) ((float4*)p)[i] = make_float4(0.f, 0.f, 0.f, 0.f);
}

// ------------------------------ encoders -----------------------------------

__global__ __launch_bounds__(256) void encode_node_kernel(
    const __bf16* __restrict__ WT0, const __bf16* __restrict__ WT1,
    const __bf16* __restrict__ WT2, const float* __restrict__ b0,
    const float* __restrict__ b1, const float* __restrict__ b2,
    const float* __restrict__ lng, const float* __restrict__ lnb,
    const float* __restrict__ velocity, const int* __restrict__ node_type,
    const float* __restrict__ nmean, const float* __restrict__ nstd,
    __bf16* __restrict__ vout) {
  __shared__ __align__(16) __bf16 X[16][40];
  __shared__ __align__(16) __bf16 Ha[16][264];
  __shared__ __align__(16) __bf16 Hb[16][264];
  __shared__ float F[16][257];
  int t = threadIdx.x, r = t >> 4, q = t & 15;
  size_t g = (size_t)blockIdx.x * 16 + r;
  if (q == 0) {
    float f[11];
    f[0] = velocity[g * 2]; f[1] = velocity[g * 2 + 1];
    int nt = node_type[g];
#pragma unroll
    for (int i = 0; i < 9; ++i) f[2 + i] = (i == nt) ? 1.f : 0.f;
#pragma unroll
    for (int i = 0; i < 11; ++i) X[r][i] = (__bf16)((f[i] - nmean[i]) / nstd[i]);
#pragma unroll
    for (int i = 11; i < 32; ++i) X[r][i] = (__bf16)0.f;
  }
  __syncthreads();
  v8f a0, a1;
  gemm16(&X[0][0], 40, 32, WT0, a0, a1);
  store_relu_bf16(a0, a1, b0, &Ha[0][0], 264);
  __syncthreads();
  gemm16(&Ha[0][0], 264, 256, WT1, a0, a1);
  store_relu_bf16(a0, a1, b1, &Hb[0][0], 264);
  __syncthreads();
  gemm16(&Hb[0][0], 264, 256, WT2, a0, a1);
  store_f32(a0, a1, b2, &F[0][0], 257);
  __syncthreads();
  float mu, rstd;
  ln_stats(&F[r][0], q, mu, rstd);
#pragma unroll
  for (int j = 0; j < 16; ++j) {
    int c = q * 16 + j;
    vout[g * 256 + c] = (__bf16)((F[r][c] - mu) * rstd * lng[c] + lnb[c]);
  }
}

__global__ __launch_bounds__(256) void encode_edge_kernel(
    const __bf16* __restrict__ WT0, const __bf16* __restrict__ WT1,
    const __bf16* __restrict__ WT2, const float* __restrict__ b0,
    const float* __restrict__ b1, const float* __restrict__ b2,
    const float* __restrict__ lng, const float* __restrict__ lnb,
    const float* __restrict__ mesh_pos, const int* __restrict__ senders,
    const int* __restrict__ receivers, const float* __restrict__ emean,
    const float* __restrict__ estd, __bf16* __restrict__ eout) {
  __shared__ __align__(16) __bf16 X[16][40];
  __shared__ __align__(16) __bf16 Ha[16][264];
  __shared__ __align__(16) __bf16 Hb[16][264];
  __shared__ float F[16][257];
  int t = threadIdx.x, r = t >> 4, q = t & 15;
  size_t g = (size_t)blockIdx.x * 16 + r;
  if (q == 0) {
    int si = senders[g], ri = receivers[g];
    float dx = mesh_pos[(size_t)si * 2]     - mesh_pos[(size_t)ri * 2];
    float dy = mesh_pos[(size_t)si * 2 + 1] - mesh_pos[(size_t)ri * 2 + 1];
    float nr = sqrtf(dx * dx + dy * dy);
    X[r][0] = (__bf16)((dx - emean[0]) / estd[0]);
    X[r][1] = (__bf16)((dy - emean[1]) / estd[1]);
    X[r][2] = (__bf16)((nr - emean[2]) / estd[2]);
#pragma unroll
    for (int i = 3; i < 32; ++i) X[r][i] = (__bf16)0.f;
  }
  __syncthreads();
  v8f a0, a1;
  gemm16(&X[0][0], 40, 32, WT0, a0, a1);
  store_relu_bf16(a0, a1, b0, &Ha[0][0], 264);
  __syncthreads();
  gemm16(&Ha[0][0], 264, 256, WT1, a0, a1);
  store_relu_bf16(a0, a1, b1, &Hb[0][0], 264);
  __syncthreads();
  gemm16(&Hb[0][0], 264, 256, WT2, a0, a1);
  store_f32(a0, a1, b2, &F[0][0], 257);
  __syncthreads();
  float mu, rstd;
  ln_stats(&F[r][0], q, mu, rstd);
#pragma unroll
  for (int j = 0; j < 16; ++j) {
    int c = q * 16 + j;
    eout[g * 256 + c] = (__bf16)((F[r][c] - mu) * rstd * lng[c] + lnb[c]);
  }
}

// ------------------------------ processor ----------------------------------

// ef = LN(MLP([e, v[s], v[r]])) + e  (in place), agg[r] += ef
__global__ __launch_bounds__(256) void edge_step_kernel(
    const __bf16* __restrict__ WT0, const __bf16* __restrict__ WT1,
    const __bf16* __restrict__ WT2, const float* __restrict__ b0,
    const float* __restrict__ b1, const float* __restrict__ b2,
    const float* __restrict__ lng, const float* __restrict__ lnb,
    __bf16* __restrict__ e, const __bf16* __restrict__ v,
    const int* __restrict__ senders, const int* __restrict__ receivers,
    float* __restrict__ agg) {
  __shared__ __align__(16) __bf16 X[16][776];   // [e | v_s | v_r], pad 8
  __shared__ __align__(16) __bf16 Ha[16][264];
  __shared__ __align__(16) __bf16 Hb[16][264];
  __shared__ float F[16][257];
  int t = threadIdx.x, r = t >> 4, q = t & 15;
  size_t g = (size_t)blockIdx.x * 16 + r;
  int si = senders[g], ri = receivers[g];
  {  // gather: each thread copies 16 bf16 (32B) from each of the 3 segments
    const uint4* pe = (const uint4*)(e + g * 256 + q * 16);
    uint4* d0 = (uint4*)(&X[r][q * 16]);
    d0[0] = pe[0]; d0[1] = pe[1];
    const uint4* ps = (const uint4*)(v + (size_t)si * 256 + q * 16);
    uint4* d1 = (uint4*)(&X[r][256 + q * 16]);
    d1[0] = ps[0]; d1[1] = ps[1];
    const uint4* pr = (const uint4*)(v + (size_t)ri * 256 + q * 16);
    uint4* d2 = (uint4*)(&X[r][512 + q * 16]);
    d2[0] = pr[0]; d2[1] = pr[1];
  }
  __syncthreads();
  v8f a0, a1;
  gemm16(&X[0][0], 776, 768, WT0, a0, a1);   // 24 k-steps
  store_relu_bf16(a0, a1, b0, &Ha[0][0], 264);
  __syncthreads();
  gemm16(&Ha[0][0], 264, 256, WT1, a0, a1);
  store_relu_bf16(a0, a1, b1, &Hb[0][0], 264);
  __syncthreads();
  gemm16(&Hb[0][0], 264, 256, WT2, a0, a1);
  store_f32(a0, a1, b2, &F[0][0], 257);
  __syncthreads();
  float mu, rstd;
  ln_stats(&F[r][0], q, mu, rstd);
#pragma unroll
  for (int j = 0; j < 16; ++j) {
    int c = q * 16 + j;
    float hv = (F[r][c] - mu) * rstd * lng[c] + lnb[c];
    float ef = hv + (float)X[r][c];          // residual: old e still in LDS
    e[g * 256 + c] = (__bf16)ef;
    atomicAdd(&agg[(size_t)ri * 256 + c], ef);
  }
}

// v = LN(MLP([v, agg])) + v (in place); zeroes agg after consuming it.
__global__ __launch_bounds__(256) void node_step_kernel(
    const __bf16* __restrict__ WT0, const __bf16* __restrict__ WT1,
    const __bf16* __restrict__ WT2, const float* __restrict__ b0,
    const float* __restrict__ b1, const float* __restrict__ b2,
    const float* __restrict__ lng, const float* __restrict__ lnb,
    __bf16* __restrict__ v, float* __restrict__ agg) {
  __shared__ __align__(16) __bf16 X[16][520];  // [v | bf16(agg)], pad 8
  __shared__ __align__(16) __bf16 Ha[16][264];
  __shared__ __align__(16) __bf16 Hb[16][264];
  __shared__ float F[16][257];
  int t = threadIdx.x, r = t >> 4, q = t & 15;
  size_t g = (size_t)blockIdx.x * 16 + r;
  {
    const uint4* pv = (const uint4*)(v + g * 256 + q * 16);
    uint4* d0 = (uint4*)(&X[r][q * 16]);
    d0[0] = pv[0]; d0[1] = pv[1];
#pragma unroll
    for (int j = 0; j < 16; ++j) {
      int c = q * 16 + j;
      X[r][256 + c] = (__bf16)agg[g * 256 + c];
      agg[g * 256 + c] = 0.f;  // ready for the next step (this row is ours)
    }
  }
  __syncthreads();
  v8f a0, a1;
  gemm16(&X[0][0], 520, 512, WT0, a0, a1);   // 16 k-steps
  store_relu_bf16(a0, a1, b0, &Ha[0][0], 264);
  __syncthreads();
  gemm16(&Ha[0][0], 264, 256, WT1, a0, a1);
  store_relu_bf16(a0, a1, b1, &Hb[0][0], 264);
  __syncthreads();
  gemm16(&Hb[0][0], 264, 256, WT2, a0, a1);
  store_f32(a0, a1, b2, &F[0][0], 257);
  __syncthreads();
  float mu, rstd;
  ln_stats(&F[r][0], q, mu, rstd);
#pragma unroll
  for (int j = 0; j < 16; ++j) {
    int c = q * 16 + j;
    float hv = (F[r][c] - mu) * rstd * lng[c] + lnb[c];
    v[g * 256 + c] = (__bf16)(hv + (float)X[r][c]);  // residual from LDS
  }
}

// ------------------------------ decoder ------------------------------------

__global__ __launch_bounds__(256) void decode_kernel(
    const __bf16* __restrict__ WT0, const __bf16* __restrict__ WT1,
    const float* __restrict__ b0, const float* __restrict__ b1,
    const float* __restrict__ W2, const float* __restrict__ b2,
    const __bf16* __restrict__ v, float* __restrict__ out) {
  __shared__ __align__(16) __bf16 X[16][264];
  __shared__ __align__(16) __bf16 Ha[16][264];
  __shared__ __align__(16) __bf16 Hb[16][264];
  int t = threadIdx.x, r = t >> 4, q = t & 15;
  size_t g = (size_t)blockIdx.x * 16 + r;
  {
    const uint4* pv = (const uint4*)(v + g * 256 + q * 16);
    uint4* d0 = (uint4*)(&X[r][q * 16]);
    d0[0] = pv[0]; d0[1] = pv[1];
  }
  __syncthreads();
  v8f a0, a1;
  gemm16(&X[0][0], 264, 256, WT0, a0, a1);
  store_relu_bf16(a0, a1, b0, &Ha[0][0], 264);
  __syncthreads();
  gemm16(&Ha[0][0], 264, 256, WT1, a0, a1);
  store_relu_bf16(a0, a1, b1, &Hb[0][0], 264);
  __syncthreads();
  if (t < 32) {  // final 256 -> 2 layer, scalar
    int rr = t >> 1, o = t & 1;
    float s = b2[o];
    for (int k = 0; k < 256; ++k) s += (float)Hb[rr][k] * W2[k * 2 + o];
    out[((size_t)blockIdx.x * 16 + rr) * 2 + o] = s;
  }
}

// ------------------------------ launcher -----------------------------------

extern "C" void kernel_launch(void* const* d_in, const int* in_sizes, int n_in,
                              void* d_out, int out_size, void* d_ws,
                              size_t ws_size, hipStream_t stream) {
  (void)n_in; (void)out_size; (void)ws_size;
  const float* velocity  = (const float*)d_in[0];
  const float* mesh_pos  = (const float*)d_in[1];
  const int*   node_type = (const int*)d_in[2];
  const int*   senders   = (const int*)d_in[3];
  const int*   receivers = (const int*)d_in[4];
  const float* node_mean = (const float*)d_in[5];
  const float* node_std  = (const float*)d_in[6];
  const float* edge_mean = (const float*)d_in[7];
  const float* edge_std  = (const float*)d_in[8];
  const float* en_W0 = (const float*)d_in[9];
  const float* en_b0 = (const float*)d_in[10];
  const float* en_W1 = (const float*)d_in[11];
  const float* en_b1 = (const float*)d_in[12];
  const float* en_W2 = (const float*)d_in[13];
  const float* en_b2 = (const float*)d_in[14];
  const float* en_g  = (const float*)d_in[15];
  const float* en_bb = (const float*)d_in[16];
  const float* ee_W0 = (const float*)d_in[17];
  const float* ee_b0 = (const float*)d_in[18];
  const float* ee_W1 = (const float*)d_in[19];
  const float* ee_b1 = (const float*)d_in[20];
  const float* ee_W2 = (const float*)d_in[21];
  const float* ee_b2 = (const float*)d_in[22];
  const float* ee_g  = (const float*)d_in[23];
  const float* ee_bb = (const float*)d_in[24];
  const float* pe_W0 = (const float*)d_in[25];   // [15,768,256]
  const float* pe_b0 = (const float*)d_in[26];   // [15,256]
  const float* pe_W1 = (const float*)d_in[27];
  const float* pe_b1 = (const float*)d_in[28];
  const float* pe_W2 = (const float*)d_in[29];
  const float* pe_b2 = (const float*)d_in[30];
  const float* pe_g  = (const float*)d_in[31];
  const float* pe_bb = (const float*)d_in[32];
  const float* pn_W0 = (const float*)d_in[33];   // [15,512,256]
  const float* pn_b0 = (const float*)d_in[34];
  const float* pn_W1 = (const float*)d_in[35];
  const float* pn_b1 = (const float*)d_in[36];
  const float* pn_W2 = (const float*)d_in[37];
  const float* pn_b2 = (const float*)d_in[38];
  const float* pn_g  = (const float*)d_in[39];
  const float* pn_bb = (const float*)d_in[40];
  const float* de_W0 = (const float*)d_in[41];
  const float* de_b0 = (const float*)d_in[42];
  const float* de_W1 = (const float*)d_in[43];
  const float* de_b1 = (const float*)d_in[44];
  const float* de_W2 = (const float*)d_in[45];   // [256,2], used as f32
  const float* de_b2 = (const float*)d_in[46];

  const int N = in_sizes[0] / 2;   // 50000
  const int E = in_sizes[3];       // 300000

  // workspace bump allocator (256B aligned)
  size_t off = 0;
  auto alloc = [&](size_t bytes) -> void* {
    off = (off + 255) & ~(size_t)255;
    void* p = (char*)d_ws + off;
    off += bytes;
    return p;
  };
  __bf16* v_bf = (__bf16*)alloc((size_t)N * 256 * 2);
  __bf16* e_bf = (__bf16*)alloc((size_t)E * 256 * 2);
  float*  agg  = (float*)alloc((size_t)N * 256 * 4);
  __bf16* wt_en0 = (__bf16*)alloc(256 * 32 * 2);
  __bf16* wt_en1 = (__bf16*)alloc(256 * 256 * 2);
  __bf16* wt_en2 = (__bf16*)alloc(256 * 256 * 2);
  __bf16* wt_ee0 = (__bf16*)alloc(256 * 32 * 2);
  __bf16* wt_ee1 = (__bf16*)alloc(256 * 256 * 2);
  __bf16* wt_ee2 = (__bf16*)alloc(256 * 256 * 2);
  __bf16* wt_pe0 = (__bf16*)alloc((size_t)15 * 256 * 768 * 2);
  __bf16* wt_pe1 = (__bf16*)alloc((size_t)15 * 256 * 256 * 2);
  __bf16* wt_pe2 = (__bf16*)alloc((size_t)15 * 256 * 256 * 2);
  __bf16* wt_pn0 = (__bf16*)alloc((size_t)15 * 256 * 512 * 2);
  __bf16* wt_pn1 = (__bf16*)alloc((size_t)15 * 256 * 256 * 2);
  __bf16* wt_pn2 = (__bf16*)alloc((size_t)15 * 256 * 256 * 2);
  __bf16* wt_de0 = (__bf16*)alloc(256 * 256 * 2);
  __bf16* wt_de1 = (__bf16*)alloc(256 * 256 * 2);

  dim3 blk(256);
  // weight conversion: f32 [K,256] -> bf16 [256,Kpad]
  prep_wt<<<dim3(32, 1), blk, 0, stream>>>(en_W0, wt_en0, 11, 32);
  prep_wt<<<dim3(256, 1), blk, 0, stream>>>(en_W1, wt_en1, 256, 256);
  prep_wt<<<dim3(256, 1), blk, 0, stream>>>(en_W2, wt_en2, 256, 256);
  prep_wt<<<dim3(32, 1), blk, 0, stream>>>(ee_W0, wt_ee0, 3, 32);
  prep_wt<<<dim3(256, 1), blk, 0, stream>>>(ee_W1, wt_ee1, 256, 256);
  prep_wt<<<dim3(256, 1), blk, 0, stream>>>(ee_W2, wt_ee2, 256, 256);
  prep_wt<<<dim3(256, 15), blk, 0, stream>>>(pe_W0, wt_pe0, 768, 768);
  prep_wt<<<dim3(256, 15), blk, 0, stream>>>(pe_W1, wt_pe1, 256, 256);
  prep_wt<<<dim3(256, 15), blk, 0, stream>>>(pe_W2, wt_pe2, 256, 256);
  prep_wt<<<dim3(256, 15), blk, 0, stream>>>(pn_W0, wt_pn0, 512, 512);
  prep_wt<<<dim3(256, 15), blk, 0, stream>>>(pn_W1, wt_pn1, 256, 256);
  prep_wt<<<dim3(256, 15), blk, 0, stream>>>(pn_W2, wt_pn2, 256, 256);
  prep_wt<<<dim3(256, 1), blk, 0, stream>>>(de_W0, wt_de0, 256, 256);
  prep_wt<<<dim3(256, 1), blk, 0, stream>>>(de_W1, wt_de1, 256, 256);

  int n4 = (N * 256) / 4;
  zero_f32<<<(n4 + 255) / 256, blk, 0, stream>>>(agg, n4);

  encode_node_kernel<<<N / 16, blk, 0, stream>>>(
      wt_en0, wt_en1, wt_en2, en_b0, en_b1, en_b2, en_g, en_bb,
      velocity, node_type, node_mean, node_std, v_bf);
  encode_edge_kernel<<<E / 16, blk, 0, stream>>>(
      wt_ee0, wt_ee1, wt_ee2, ee_b0, ee_b1, ee_b2, ee_g, ee_bb,
      mesh_pos, senders, receivers, edge_mean, edge_std, e_bf);

  for (int s = 0; s < 15; ++s) {
    edge_step_kernel<<<E / 16, blk, 0, stream>>>(
        wt_pe0 + (size_t)s * 256 * 768, wt_pe1 + (size_t)s * 256 * 256,
        wt_pe2 + (size_t)s * 256 * 256, pe_b0 + s * 256, pe_b1 + s * 256,
        pe_b2 + s * 256, pe_g + s * 256, pe_bb + s * 256,
        e_bf, v_bf, senders, receivers, agg);
    node_step_kernel<<<N / 16, blk, 0, stream>>>(
        wt_pn0 + (size_t)s * 256 * 512, wt_pn1 + (size_t)s * 256 * 256,
        wt_pn2 + (size_t)s * 256 * 256, pn_b0 + s * 256, pn_b1 + s * 256,
        pn_b2 + s * 256, pn_g + s * 256, pn_bb + s * 256, v_bf, agg);
  }

  decode_kernel<<<N / 16, blk, 0, stream>>>(
      wt_de0, wt_de1, de_b0, de_b1, de_W2, de_b2, v_bf, (float*)d_out);
}
